// SortPool_87729001988951
// MI455X (gfx1250) — matrix operand order repaired and checked by reading, not compiled, not run on hardware
//
#include <hip/hip_runtime.h>
#include <stdint.h>
#include <math.h>

#define N_NODESC  100000
#define N_EDGESC  1600000
#define N_GRAPHSC 2000
#define HIDDENC   128
#define NLAYERSC  4
#define NCLASSESC 10
#define K_POOL    30
#define CONV_OUTC 32
#define KSIZEC    5
#define CONV_T    (K_POOL - KSIZEC + 1)   /* 26 */
#define FLATC     (CONV_OUTC * CONV_T)    /* 832 */

typedef _Float16 v16h __attribute__((ext_vector_type(16)));
typedef float    v8f  __attribute__((ext_vector_type(8)));

// ---------------------------------------------------------------- utilities
__global__ void k_fill_f32(float* __restrict__ p, float v, int n) {
    int i = blockIdx.x * blockDim.x + threadIdx.x;
    if (i < n) p[i] = v;
}
__global__ void k_fill_i32(int* __restrict__ p, int v, int n) {
    int i = blockIdx.x * blockDim.x + threadIdx.x;
    if (i < n) p[i] = v;
}
__global__ void k_degree(const int* __restrict__ row, float* __restrict__ deg, int e) {
    int i = blockIdx.x * blockDim.x + threadIdx.x;
    if (i < e) atomicAdd(&deg[row[i]], 1.0f);
}
__global__ void k_rsqrt_deg(const float* __restrict__ deg, float* __restrict__ dinv, int n) {
    int i = blockIdx.x * blockDim.x + threadIdx.x;
    if (i < n) dinv[i] = rsqrtf(deg[i] + 1.0f);
}
__global__ void k_f32_to_f16(const float* __restrict__ s, _Float16* __restrict__ d, int n) {
    int i = blockIdx.x * blockDim.x + threadIdx.x;
    if (i < n) d[i] = (_Float16)s[i];
}
// fused: p = relu(p); d = f16(relu(p))   (feeds next layer's WMMA A-matrix)
__global__ void k_relu_cvt(float* __restrict__ p, _Float16* __restrict__ d, int n) {
    int i = blockIdx.x * blockDim.x + threadIdx.x;
    if (i < n) {
        float v = fmaxf(p[i], 0.0f);
        p[i] = v;
        d[i] = (_Float16)v;
    }
}
__global__ void k_relu(float* __restrict__ p, int n) {
    int i = blockIdx.x * blockDim.x + threadIdx.x;
    if (i < n) p[i] = fmaxf(p[i], 0.0f);
}
// WhT[n*128+k] = W[k*128+n]  (fp16, column-major for contiguous B-fragment loads)
__global__ void k_w_transpose_f16(const float* __restrict__ W, _Float16* __restrict__ WhT) {
    int i = blockIdx.x * blockDim.x + threadIdx.x;
    if (i < HIDDENC * HIDDENC) {
        int k = i >> 7, n = i & 127;
        WhT[(size_t)n * HIDDENC + k] = (_Float16)W[(size_t)k * HIDDENC + n];
    }
}

// ------------------------------- WMMA GEMM + fused self-loop/bias epilogue
// HW = hh @ W ;  H0 = HW * dinv^2 + bias   (H0 then receives edge atomics)
// M = 100000 -> 3125 blocks x 2 M-tiles; 8 waves/block = N-tiles; K = 128.
__global__ __launch_bounds__(256)
void k_gemm_wmma_fused(const _Float16* __restrict__ hh, const _Float16* __restrict__ WhT,
                       const float* __restrict__ dinv, const float* __restrict__ bias,
                       float* __restrict__ hw, float* __restrict__ hout) {
    const int lane = threadIdx.x & 31;
    const int wave = threadIdx.x >> 5;     // N-tile 0..7
    const int mt0  = blockIdx.x * 2;       // first of two M-tiles
    const int l15  = lane & 15;
    const int kh   = lane >> 4;            // K-half select (contiguous A/B k order)

    const _Float16* __restrict__ a0   = hh + (size_t)(mt0 * 16 + l15) * HIDDENC;
    const _Float16* __restrict__ a1   = a0 + (size_t)16 * HIDDENC;
    const _Float16* __restrict__ bcol = WhT + (size_t)(wave * 16 + l15) * HIDDENC + kh * 16;

    v8f acc[2] = {{}, {}};
#pragma unroll
    for (int kk = 0; kk < 4; ++kk) {
        const int kbase = kk * 32;
        v16h A0, A1;
#pragma unroll
        for (int i = 0; i < 8; ++i) {
            // A 16x32, contiguous k order as in B: VGPR i holds K = kh*16 + i*2 (+1)
            int k = kbase + kh * 16 + (i << 1);
            A0[2 * i]     = a0[k];
            A0[2 * i + 1] = a0[k + 1];
            A1[2 * i]     = a1[k];
            A1[2 * i + 1] = a1[k + 1];
        }
        // B 32x16: lane-half kh supplies K = kbase + kh*16 .. +15, contiguous in WhT
        v16h B = *(const v16h*)(bcol + kbase);
        acc[0] = __builtin_amdgcn_wmma_f32_16x16x32_f16(false, A0, false, B,
                                                        (short)0, acc[0], false, false);
        acc[1] = __builtin_amdgcn_wmma_f32_16x16x32_f16(false, A1, false, B,
                                                        (short)0, acc[1], false, false);
    }
    // C/D layout: n = lane&15 ; rows m = kh*8 + r
    const int   n  = wave * 16 + l15;
    const float bn = bias[n];
#pragma unroll
    for (int t = 0; t < 2; ++t) {
        const int m0 = (mt0 + t) * 16 + kh * 8;
#pragma unroll
        for (int r = 0; r < 8; ++r) {
            const int   m = m0 + r;
            const float v = acc[t][r];
            const float d = dinv[m];            // wave-uniform per row: broadcast hit
            hw[(size_t)m * HIDDENC + n]   = v;
            hout[(size_t)m * HIDDENC + n] = v * d * d + bn;
        }
    }
}

// --------------------------------------------- GCN edge aggregation (atomics)
__global__ void k_edge_scatter(const int* __restrict__ row, const int* __restrict__ col,
                               const float* __restrict__ dinv, const float* __restrict__ hw,
                               float* __restrict__ hout) {
    long tid = (long)blockIdx.x * blockDim.x + threadIdx.x;
    if (tid >= (long)N_EDGESC * 32) return;
    int e = (int)(tid >> 5);
    int q = (int)(tid & 31);
    int r = row[e], c = col[e];
    float nrm = dinv[r] * dinv[c];
    const float4 v = *(const float4*)(hw + (size_t)c * HIDDENC + q * 4);
    float* dst = hout + (size_t)r * HIDDENC + q * 4;
    atomicAdd(dst + 0, v.x * nrm);
    atomicAdd(dst + 1, v.y * nrm);
    atomicAdd(dst + 2, v.z * nrm);
    atomicAdd(dst + 3, v.w * nrm);
}

// ----------------------------------------------------------------- SortPool
__global__ void k_counts(const int* __restrict__ batch, int* __restrict__ counts, int n) {
    int i = blockIdx.x * blockDim.x + threadIdx.x;
    if (i < n) atomicAdd(&counts[batch[i]], 1);
}
__global__ void k_starts(const int* __restrict__ counts, int* __restrict__ starts) {
    if (blockIdx.x == 0 && threadIdx.x == 0) {
        int run = 0;
        for (int g = 0; g < N_GRAPHSC; ++g) { starts[g] = run; run += counts[g]; }
    }
}
// stable top-K selection per graph (batch is sorted -> graphs contiguous)
__global__ void k_select(const float* __restrict__ h, const int* __restrict__ counts,
                         const int* __restrict__ starts, int* __restrict__ sel) {
    int g = blockIdx.x * blockDim.x + threadIdx.x;
    if (g >= N_GRAPHSC) return;
    int cnt = counts[g], s = starts[g];
    float lastKey = __builtin_inff();
    int   lastIdx = -1;
    for (int k = 0; k < K_POOL; ++k) {
        int best = -1; float bestKey = 0.0f;
        for (int j = 0; j < cnt; ++j) {
            int idx = s + j;
            float key = h[(size_t)idx * HIDDENC + (HIDDENC - 1)];
            bool elig = (key < lastKey) || (key == lastKey && idx > lastIdx);
            if (elig && (best < 0 || key > bestKey)) { best = idx; bestKey = key; }
        }
        sel[g * K_POOL + k] = best;
        if (best >= 0) { lastKey = bestKey; lastIdx = best; }
    }
}
__global__ void k_gather_pool(const float* __restrict__ h, const int* __restrict__ sel,
                              float* __restrict__ pooled) {
    int gk = blockIdx.x;          // 0 .. N_GRAPHS*K_POOL-1
    int f  = threadIdx.x;         // 0 .. 127
    int idx = sel[gk];
    pooled[(size_t)gk * HIDDENC + f] = (idx >= 0) ? h[(size_t)idx * HIDDENC + f] : 0.0f;
}

// ------------------------------------------------------------ Conv1d + MLP
__global__ void k_conv(const float* __restrict__ pooled, const float* __restrict__ cw,
                       const float* __restrict__ cb, float* __restrict__ convz) {
    int tid = blockIdx.x * blockDim.x + threadIdx.x;
    if (tid >= N_GRAPHSC * CONV_OUTC * CONV_T) return;
    int t = tid % CONV_T;
    int o = (tid / CONV_T) % CONV_OUTC;
    int g = tid / (CONV_T * CONV_OUTC);
    float acc = cb[o];
    const float* P = pooled + (size_t)g * K_POOL * HIDDENC + (size_t)t * HIDDENC;
    const float* W = cw + (size_t)o * HIDDENC * KSIZEC;
#pragma unroll
    for (int s = 0; s < KSIZEC; ++s)
        for (int c = 0; c < HIDDENC; ++c)
            acc += P[s * HIDDENC + c] * W[c * KSIZEC + s];
    convz[tid] = fmaxf(acc, 0.0f);   // flatten order = g*(32*26) + o*26 + t
}
__global__ void k_lin1(const float* __restrict__ z, const float* __restrict__ w,
                       const float* __restrict__ b, float* __restrict__ z2) {
    int tid = blockIdx.x * blockDim.x + threadIdx.x;
    if (tid >= N_GRAPHSC * HIDDENC) return;
    int j = tid & 127, g = tid >> 7;
    float acc = b[j];
    const float* Z = z + (size_t)g * FLATC;
    for (int i = 0; i < FLATC; ++i)
        acc += Z[i] * w[(size_t)i * HIDDENC + j];
    z2[tid] = fmaxf(acc, 0.0f);
}
__global__ void k_lin2_lsm(const float* __restrict__ z2, const float* __restrict__ w,
                           const float* __restrict__ b, float* __restrict__ out) {
    int g = blockIdx.x * blockDim.x + threadIdx.x;
    if (g >= N_GRAPHSC) return;
    float l[NCLASSESC];
#pragma unroll
    for (int c = 0; c < NCLASSESC; ++c) l[c] = b[c];
    for (int j = 0; j < HIDDENC; ++j) {
        float v = z2[(size_t)g * HIDDENC + j];
#pragma unroll
        for (int c = 0; c < NCLASSESC; ++c) l[c] += v * w[j * NCLASSESC + c];
    }
    float m = l[0];
#pragma unroll
    for (int c = 1; c < NCLASSESC; ++c) m = fmaxf(m, l[c]);
    float sum = 0.0f;
#pragma unroll
    for (int c = 0; c < NCLASSESC; ++c) sum += expf(l[c] - m);
    float lse = logf(sum);
#pragma unroll
    for (int c = 0; c < NCLASSESC; ++c) out[(size_t)g * NCLASSESC + c] = l[c] - m - lse;
}

// ------------------------------------------------------------------ launch
extern "C" void kernel_launch(void* const* d_in, const int* in_sizes, int n_in,
                              void* d_out, int out_size, void* d_ws, size_t ws_size,
                              hipStream_t stream) {
    const float* x     = (const float*)d_in[0];
    const int*   ei    = (const int*)  d_in[1];
    const int*   batch = (const int*)  d_in[2];
    const float* W1    = (const float*)d_in[3];
    const float* b1    = (const float*)d_in[4];
    const float* Wsm   = (const float*)d_in[5];
    const float* bsm   = (const float*)d_in[6];
    const float* cw    = (const float*)d_in[7];
    const float* cb    = (const float*)d_in[8];
    const float* l1w   = (const float*)d_in[9];
    const float* l1b   = (const float*)d_in[10];
    const float* l2w   = (const float*)d_in[11];
    const float* l2b   = (const float*)d_in[12];
    float* out = (float*)d_out;

    const int* row = ei;
    const int* col = ei + N_EDGESC;

    // workspace carve-out (256B aligned)
    char* ws = (char*)d_ws;
    size_t off = 0;
    auto carve = [&](size_t bytes) -> char* {
        char* p = ws + off;
        off = (off + bytes + 255) & ~(size_t)255;
        return p;
    };
    float*     deg    = (float*)    carve((size_t)N_NODESC * 4);
    float*     dinv   = (float*)    carve((size_t)N_NODESC * 4);
    float*     H0     = (float*)    carve((size_t)N_NODESC * HIDDENC * 4);
    float*     HW     = (float*)    carve((size_t)N_NODESC * HIDDENC * 4);
    _Float16*  hh     = (_Float16*) carve((size_t)N_NODESC * HIDDENC * 2);
    _Float16*  WhT    = (_Float16*) carve((size_t)HIDDENC * HIDDENC * 2);
    int*       counts = (int*)      carve((size_t)N_GRAPHSC * 4);
    int*       starts = (int*)      carve((size_t)N_GRAPHSC * 4);
    int*       sel    = (int*)      carve((size_t)N_GRAPHSC * K_POOL * 4);

    // post-GCN buffers alias HW (free once layers finish); H0 stays live for pooling
    size_t pooledB = (size_t)N_GRAPHSC * K_POOL * HIDDENC * 4;  // 30.72 MB
    size_t convzB  = (size_t)N_GRAPHSC * FLATC * 4;             //  6.66 MB
    float* pooled = HW;
    float* convz  = (float*)((char*)HW + ((pooledB + 255) & ~(size_t)255));
    float* z2     = (float*)((char*)convz + ((convzB + 255) & ~(size_t)255));

    const int NT = 256;
    const int NH = N_NODESC * HIDDENC;  // 12.8M

    // degree + normalization
    k_fill_f32<<<(N_NODESC + NT - 1) / NT, NT, 0, stream>>>(deg, 0.0f, N_NODESC);
    k_degree<<<(N_EDGESC + NT - 1) / NT, NT, 0, stream>>>(row, deg, N_EDGESC);
    k_rsqrt_deg<<<(N_NODESC + NT - 1) / NT, NT, 0, stream>>>(deg, dinv, N_NODESC);

    // 4 GCN layers (hh = fp16 snapshot of layer input)
    k_f32_to_f16<<<(NH + NT - 1) / NT, NT, 0, stream>>>(x, hh, NH);
    for (int l = 0; l < NLAYERSC; ++l) {
        const float* W    = (l == 0) ? W1 : Wsm + (size_t)(l - 1) * HIDDENC * HIDDENC;
        const float* bias = (l == 0) ? b1 : bsm + (size_t)(l - 1) * HIDDENC;

        k_w_transpose_f16<<<(HIDDENC * HIDDENC + NT - 1) / NT, NT, 0, stream>>>(W, WhT);
        k_gemm_wmma_fused<<<N_NODESC / 32, 256, 0, stream>>>(hh, WhT, dinv, bias, HW, H0);
        long sth = (long)N_EDGESC * 32;
        k_edge_scatter<<<(unsigned)((sth + NT - 1) / NT), NT, 0, stream>>>(row, col, dinv, HW, H0);
        if (l + 1 < NLAYERSC)
            k_relu_cvt<<<(NH + NT - 1) / NT, NT, 0, stream>>>(H0, hh, NH);
        else
            k_relu<<<(NH + NT - 1) / NT, NT, 0, stream>>>(H0, NH);
    }

    // SortPool
    k_fill_i32<<<(N_GRAPHSC + NT - 1) / NT, NT, 0, stream>>>(counts, 0, N_GRAPHSC);
    k_counts<<<(N_NODESC + NT - 1) / NT, NT, 0, stream>>>(batch, counts, N_NODESC);
    k_starts<<<1, 1, 0, stream>>>(counts, starts);
    k_select<<<(N_GRAPHSC + NT - 1) / NT, NT, 0, stream>>>(H0, counts, starts, sel);
    k_gather_pool<<<N_GRAPHSC * K_POOL, HIDDENC, 0, stream>>>(H0, sel, pooled);

    // Conv1d + MLP head
    int convN = N_GRAPHSC * CONV_OUTC * CONV_T;
    k_conv<<<(convN + NT - 1) / NT, NT, 0, stream>>>(pooled, cw, cb, convz);
    k_lin1<<<(N_GRAPHSC * HIDDENC + NT - 1) / NT, NT, 0, stream>>>(convz, l1w, l1b, z2);
    k_lin2_lsm<<<(N_GRAPHSC + NT - 1) / NT, NT, 0, stream>>>(z2, l2w, l2b, out);

    (void)in_sizes; (void)n_in; (void)out_size; (void)ws_size;
}
